// RetAugmentationLinearAttention_146028888567
// MI455X (gfx1250) — compile-verified, matrix operand order checked by reading, not run on hardware
//
#include <hip/hip_runtime.h>
#include <hip/hip_bf16.h>
#include <stdint.h>

// ---------------------------------------------------------------------------
// Types for CDNA5 WMMA (wave32): v_wmma_f32_16x16x32_bf16
// ---------------------------------------------------------------------------
typedef __attribute__((ext_vector_type(16))) __bf16         v16bf;
typedef __attribute__((ext_vector_type(8)))  float          v8f;
typedef __attribute__((ext_vector_type(8)))  unsigned short ushort8;

union FragBF {
    v16bf   bf;
    ushort8 u[2];
};

__device__ __forceinline__ unsigned short f32_to_bf16_bits(float f) {
    union { float f; unsigned int u; } v;
    v.f = f;
    unsigned int r = v.u + 0x7FFFu + ((v.u >> 16) & 1u);  // round-to-nearest-even
    return (unsigned short)(r >> 16);
}

// CDNA5 async copy: 16B per lane, global -> LDS, tracked by ASYNCcnt.
__device__ __forceinline__ void async_copy_b128(unsigned lds_addr, const void* gaddr) {
    asm volatile("global_load_async_to_lds_b128 %0, %1, off"
                 :: "v"(lds_addr), "v"(gaddr)
                 : "memory");
}

__device__ __forceinline__ void wait_async0() {
#if __has_builtin(__builtin_amdgcn_s_wait_asynccnt)
    __builtin_amdgcn_s_wait_asynccnt(0);
#else
    asm volatile("s_wait_asynccnt 0x0" ::: "memory");
#endif
}

// ---------------------------------------------------------------------------
// Generic batched/strided bf16 WMMA GEMM (all operands unit K-stride).
//   C(b,m,n) = sum_k A(b,m,k)*B(b,n,k)  (+ optional fp32 residual)
//   A(b,m,k) = A[b*sAb + m*sAm + k]
//   B(b,n,k) = B[b*sBb + n*sBn + k]
//   C(b,m,n) = C[b*sCb + m*sCm + n*sCn]   (generic strides -> free transpose)
// ASYNC_A/ASYNC_B: bf16 operand staged with global_load_async_to_lds_b128
// (double-buffered pipeline). fp32 operands: global_load_b128 + cvt + ds_store.
// Tile: 128x128x32, 256 threads = 8 wave32 waves (2x4), 8 WMMA/wave/K-step.
// Requires M%128==0, N%128==0, K%32==0, rows 16B-aligned (true below).
// ---------------------------------------------------------------------------
constexpr int LDSS = 40;               // 32 bf16 + 8 pad: 80B rows (16B aligned)
constexpr int TILE_US = 128 * LDSS;    // 5120 ushorts = 10 KB per tile

template <typename TA, typename TB, typename TC,
          bool ASYNC_A, bool ASYNC_B, bool RESID>
__global__ __launch_bounds__(256)
void wmma_gemm_kernel(const TA* __restrict__ A, long sAm, long sAb,
                      const TB* __restrict__ B, long sBn, long sBb,
                      TC* __restrict__ C, long sCm, long sCn, long sCb,
                      const float* __restrict__ R, long sRb,
                      int Mdim, int Ndim, int Kdim)
{
    static_assert(!ASYNC_A || sizeof(TA) == 2, "async A operand must be bf16");
    static_assert(!ASYNC_B || sizeof(TB) == 2, "async B operand must be bf16");

    __shared__ __align__(16) unsigned short Atile[2][TILE_US];
    __shared__ __align__(16) unsigned short Btile[2][TILE_US];

    const int bz = blockIdx.z;
    A += (long)bz * sAb;
    B += (long)bz * sBb;
    C += (long)bz * sCb;
    const float* Rb = RESID ? (R + (long)bz * sRb) : nullptr;

    const int m0 = blockIdx.y * 128;
    const int n0 = blockIdx.x * 128;

    const int tid   = threadIdx.x;
    const int wave  = tid >> 5;       // 0..7 (wave32)
    const int lane  = tid & 31;
    const int waveM = wave >> 2;      // 0..1 -> 64 rows
    const int waveN = wave & 3;       // 0..3 -> 32 cols
    const int lrow  = lane & 15;
    const int lhalf = lane >> 4;

    // convert-path staging: thread -> (row 0..127, k-halfslice 0/16)
    const int srow = tid >> 1;
    const int scol = (tid & 1) * 16;

    v8f acc[4][2];
    const v8f vzero = {0.f, 0.f, 0.f, 0.f, 0.f, 0.f, 0.f, 0.f};
#pragma unroll
    for (int mi = 0; mi < 4; ++mi)
#pragma unroll
        for (int ni = 0; ni < 2; ++ni) acc[mi][ni] = vzero;

    // convert-path: 16 elems (4x global_load_b128) -> bf16 -> 2x ds_store_b128
    auto stage_cvt = [&](unsigned short* dst, const void* srcv, bool pf) {
        unsigned short tmp[16];
        if constexpr (sizeof(TA) == 4 || sizeof(TB) == 4) { /* keep type below */ }
        (void)pf;
        const float4* s4 = (const float4*)srcv;
#pragma unroll
        for (int j = 0; j < 4; ++j) {
            const float4 f = s4[j];
            tmp[4 * j + 0] = f32_to_bf16_bits(f.x);
            tmp[4 * j + 1] = f32_to_bf16_bits(f.y);
            tmp[4 * j + 2] = f32_to_bf16_bits(f.z);
            tmp[4 * j + 3] = f32_to_bf16_bits(f.w);
        }
        *(ushort8*)(dst + 0) = *(const ushort8*)&tmp[0];
        *(ushort8*)(dst + 8) = *(const ushort8*)&tmp[8];
    };

    // ---- stage tile (k0..k0+31) into buffer `buf`
    auto stage = [&](int buf, int k0) {
        if constexpr (ASYNC_A) {
#pragma unroll
            for (int ch = 0; ch < 2; ++ch) {
                const int c    = tid + ch * 256;      // 512 chunks of 16B
                const int row  = c >> 2;
                const int part = c & 3;
                const unsigned lds =
                    (unsigned)(size_t)(&Atile[buf][row * LDSS + part * 8]);
                const char* g = (const char*)A +
                    (((long)(m0 + row) * sAm + k0) << 1) + part * 16;
                async_copy_b128(lds, g);
            }
        } else {
            const TA* rowp = A + (long)(m0 + srow) * sAm + k0 + scol;
            if (k0 + 32 < Kdim) __builtin_prefetch(rowp + 32, 0, 1);  // prefetch_b8
            stage_cvt(&Atile[buf][srow * LDSS + scol], rowp, false);
        }
        if constexpr (ASYNC_B) {
#pragma unroll
            for (int ch = 0; ch < 2; ++ch) {
                const int c    = tid + ch * 256;
                const int row  = c >> 2;
                const int part = c & 3;
                const unsigned lds =
                    (unsigned)(size_t)(&Btile[buf][row * LDSS + part * 8]);
                const char* g = (const char*)B +
                    (((long)(n0 + row) * sBn + k0) << 1) + part * 16;
                async_copy_b128(lds, g);
            }
        } else {
            const TB* rowp = B + (long)(n0 + srow) * sBn + k0 + scol;
            if (k0 + 32 < Kdim) __builtin_prefetch(rowp + 32, 0, 1);
            stage_cvt(&Btile[buf][srow * LDSS + scol], rowp, false);
        }
    };

    // ---- 8 WMMAs per wave from buffer `buf` (2x ds_load_b128 per fragment)
    auto compute = [&](int buf) {
        FragBF a[4], b[2];
#pragma unroll
        for (int mi = 0; mi < 4; ++mi) {
            const unsigned short* p =
                &Atile[buf][(waveM * 64 + mi * 16 + lrow) * LDSS + 8 * lhalf];
            a[mi].u[0] = *(const ushort8*)p;
            a[mi].u[1] = *(const ushort8*)(p + 16);
        }
#pragma unroll
        for (int ni = 0; ni < 2; ++ni) {
            const unsigned short* p =
                &Btile[buf][(waveN * 32 + ni * 16 + lrow) * LDSS + 8 * lhalf];
            b[ni].u[0] = *(const ushort8*)p;
            b[ni].u[1] = *(const ushort8*)(p + 16);
        }
#pragma unroll
        for (int mi = 0; mi < 4; ++mi)
#pragma unroll
            for (int ni = 0; ni < 2; ++ni)
                acc[mi][ni] = __builtin_amdgcn_wmma_f32_16x16x32_bf16(
                    false, a[mi].bf, false, b[ni].bf,
                    (short)0, acc[mi][ni], false, false);
    };

    // ---- double-buffered pipeline: copy tile i+1 while computing tile i
    const int T = Kdim >> 5;
    stage(0, 0);
    if constexpr (ASYNC_A || ASYNC_B) wait_async0();
    __syncthreads();
    for (int i = 0; i < T; ++i) {
        if (i + 1 < T) stage((i + 1) & 1, (i + 1) << 5);
        compute(i & 1);
        if constexpr (ASYNC_A || ASYNC_B) wait_async0();
        __syncthreads();
    }

    // ---- epilogue: C/D layout: VGPR v -> M = 8*half + v, N = lane&15
#pragma unroll
    for (int mi = 0; mi < 4; ++mi) {
#pragma unroll
        for (int ni = 0; ni < 2; ++ni) {
            const int rbase = m0 + waveM * 64 + mi * 16 + 8 * lhalf;
            const int col   = n0 + waveN * 32 + ni * 16 + lrow;
#pragma unroll
            for (int v = 0; v < 8; ++v) {
                const long off = (long)(rbase + v) * sCm + (long)col * sCn;
                float val = acc[mi][ni][v];
                if constexpr (RESID) val += Rb[off];
                if constexpr (sizeof(TC) == 4) {
                    ((float*)C)[off] = val;
                } else {
                    ((unsigned short*)C)[off] = f32_to_bf16_bits(val);
                }
            }
        }
    }
}

// ---------------------------------------------------------------------------
// Row softmax over 512 channels, one wave32 per row, output bf16.
// ---------------------------------------------------------------------------
__global__ __launch_bounds__(256)
void softmax512_kernel(const float* __restrict__ in,
                       unsigned short* __restrict__ out, long nrows)
{
    const long row = (long)blockIdx.x * 8 + (threadIdx.x >> 5);
    if (row >= nrows) return;
    const int lane = threadIdx.x & 31;
    const float* rp = in + row * 512;

    float v[16];
    float mx = -3.402823466e38f;
#pragma unroll
    for (int i = 0; i < 16; ++i) {
        v[i] = rp[lane + i * 32];
        mx = fmaxf(mx, v[i]);
    }
#pragma unroll
    for (int off = 16; off >= 1; off >>= 1) mx = fmaxf(mx, __shfl_xor(mx, off, 32));

    float s = 0.f;
#pragma unroll
    for (int i = 0; i < 16; ++i) {
        v[i] = __expf(v[i] - mx);
        s += v[i];
    }
#pragma unroll
    for (int off = 16; off >= 1; off >>= 1) s += __shfl_xor(s, off, 32);

    const float inv = 1.0f / s;
    unsigned short* op = out + row * 512;
#pragma unroll
    for (int i = 0; i < 16; ++i) op[lane + i * 32] = f32_to_bf16_bits(v[i] * inv);
}

// ---------------------------------------------------------------------------
// LDS-tiled 32x32 transpose: h (fp32 [B,4096,512]) -> kvT (bf16 [B,512,6144])
//   kvT[b][c][t] = bf16(h[b][t][c]), t in [0,4096)
// Coalesced reads along channels, coalesced bf16 writes along tokens.
// Columns [4096,6144) of kvT come from the W_cond GEMM (transposed store).
// ---------------------------------------------------------------------------
__global__ __launch_bounds__(256)
void pack_hT_kernel(const float* __restrict__ h, unsigned short* __restrict__ kvT)
{
    __shared__ unsigned short tile[32][33];
    const int bid  = blockIdx.x;           // B * 128 * 16 blocks
    const int cblk = bid & 15;             // 512/32
    const int tblk = (bid >> 4) & 127;     // 4096/32
    const int b    = bid >> 11;
    const int t0 = tblk * 32, c0 = cblk * 32;
    const int col  = threadIdx.x & 31;
    const int row4 = threadIdx.x >> 5;     // 0..7

#pragma unroll
    for (int j = 0; j < 4; ++j) {
        const int r = row4 * 4 + j;        // token offset
        tile[r][col] =
            f32_to_bf16_bits(h[((long)b * 4096 + t0 + r) * 512 + c0 + col]);
    }
    __syncthreads();
#pragma unroll
    for (int j = 0; j < 4; ++j) {
        const int r = row4 * 4 + j;        // channel offset
        kvT[((long)b * 512 + c0 + r) * 6144 + t0 + col] = tile[col][r];
    }
}

// ---------------------------------------------------------------------------
// Orchestration (exact algebraic restructuring of the reference):
//   kvT[:, :, N:] = (h_retrieved @ Wc^T)^T            (bf16, transposed store)
//   att = softmax(h @ Wa^T)                           (bf16)
//   G   = kv^T kv       (per batch, K=6144, symmetric)       [async x2]
//   f   = Wk G Wv^T  ==  k^T v                        (two 512^3 GEMMs)
//   gT  = (f @ Wo^T)^T                                (transposed store)
//   out = h + att @ gT^T                                      [async x2]
// ---------------------------------------------------------------------------
extern "C" void kernel_launch(void* const* d_in, const int* in_sizes, int n_in,
                              void* d_out, int out_size, void* d_ws, size_t ws_size,
                              hipStream_t stream)
{
    const float* h  = (const float*)d_in[0];
    const float* hr = (const float*)d_in[1];
    const float* Wc = (const float*)d_in[2];
    const float* Wa = (const float*)d_in[3];
    const float* Wk = (const float*)d_in[4];
    const float* Wv = (const float*)d_in[5];
    const float* Wo = (const float*)d_in[6];
    float* out = (float*)d_out;
    (void)in_sizes; (void)n_in; (void)out_size; (void)ws_size;

    constexpr long Bb = 8, N = 4096, M = 2048, IN = 512, D = 512, CTX = 768;
    constexpr long NM = N + M;  // 6144

    char* ws = (char*)d_ws;
    auto aup = [](size_t x) { return (x + 255) & ~size_t(255); };
    size_t off = 0;
    unsigned short* kvT = (unsigned short*)(ws + off); off = aup(off + (size_t)Bb * IN * NM * 2);
    float* logits       = (float*)(ws + off);          off = aup(off + (size_t)Bb * N * D * 4);
    unsigned short* at  = (unsigned short*)(ws + off); off = aup(off + (size_t)Bb * N * D * 2);
    unsigned short* G   = (unsigned short*)(ws + off); off = aup(off + (size_t)Bb * D * D * 2);
    unsigned short* P   = (unsigned short*)(ws + off); off = aup(off + (size_t)Bb * D * D * 2);
    unsigned short* F   = (unsigned short*)(ws + off); off = aup(off + (size_t)Bb * D * D * 2);
    unsigned short* gT  = (unsigned short*)(ws + off); off = aup(off + (size_t)Bb * D * IN * 2);

    // 1) h -> kvT[:, :, 0:N] as bf16 (LDS-tiled transpose)
    pack_hT_kernel<<<dim3((unsigned)(Bb * 128 * 16)), 256, 0, stream>>>(h, kvT);

    // 2) kvT[:, :, N:] = (h_retrieved @ Wc^T)^T  (batched M=2048,N=512,K=768)
    wmma_gemm_kernel<float, float, unsigned short, false, false, false>
        <<<dim3(IN / 128, M / 128, Bb), 256, 0, stream>>>(
            hr, CTX, M * CTX,
            Wc, CTX, 0,
            kvT + N, /*sCm=*/1, /*sCn=*/NM, IN * NM,
            nullptr, 0, (int)M, (int)IN, (int)CTX);

    // 3) logits = h @ Wa^T  (flat M=B*N=32768, N=512, K=512)
    wmma_gemm_kernel<float, float, float, false, false, false>
        <<<dim3(D / 128, (Bb * N) / 128, 1), 256, 0, stream>>>(
            h, IN, 0,
            Wa, IN, 0,
            logits, D, 1, 0,
            nullptr, 0, (int)(Bb * N), (int)D, (int)IN);

    // 4) attn = softmax(logits) over channels -> bf16
    softmax512_kernel<<<dim3((unsigned)((Bb * N) / 8)), 256, 0, stream>>>(
        logits, at, Bb * N);

    // 5) G = kv^T kv  (batched, M=N=512, K=6144) — both operands async bf16
    wmma_gemm_kernel<unsigned short, unsigned short, unsigned short, true, true, false>
        <<<dim3(D / 128, D / 128, Bb), 256, 0, stream>>>(
            kvT, NM, IN * NM,
            kvT, NM, IN * NM,
            G, D, 1, D * D,
            nullptr, 0, (int)D, (int)D, (int)NM);

    // 6) P = Wk @ G  (batched 512^3); G symmetric -> B read row-major, async
    wmma_gemm_kernel<float, unsigned short, unsigned short, false, true, false>
        <<<dim3(D / 128, D / 128, Bb), 256, 0, stream>>>(
            Wk, IN, 0,
            G, D, D * D,
            P, D, 1, D * D,
            nullptr, 0, (int)D, (int)D, (int)IN);

    // 7) F = P @ Wv^T == k^T v  (batched 512^3); A async
    wmma_gemm_kernel<unsigned short, float, unsigned short, true, false, false>
        <<<dim3(D / 128, D / 128, Bb), 256, 0, stream>>>(
            P, D, D * D,
            Wv, IN, 0,
            F, D, 1, D * D,
            nullptr, 0, (int)D, (int)D, (int)IN);

    // 8) gT = (F @ Wo^T)^T  (batched 512^3); A async, transposed store
    wmma_gemm_kernel<unsigned short, float, unsigned short, true, false, false>
        <<<dim3(IN / 128, D / 128, Bb), 256, 0, stream>>>(
            F, D, D * D,
            Wo, D, 0,
            gT, /*sCm=*/1, /*sCn=*/D, D * IN,
            nullptr, 0, (int)D, (int)IN, (int)D);

    // 9) out = h + attn @ g  (batched M=4096,N=512,K=512) — both operands async
    wmma_gemm_kernel<unsigned short, unsigned short, float, true, true, true>
        <<<dim3(IN / 128, N / 128, Bb), 256, 0, stream>>>(
            at, D, N * D,
            gT, D, D * IN,
            out, IN, 1, N * IN,
            h, N * IN,
            (int)N, (int)IN, (int)D);
}
